// SessionGraph_824633721638
// MI455X (gfx1250) — matrix operand order repaired and checked by reading, not compiled, not run on hardware
//
#include <hip/hip_runtime.h>

// ---------------------------------------------------------------------------
// SessionGraph (star-GAT) for MI455X / gfx1250.
// wave32 + WMMA f16->f32 + async global->LDS DMA (ASYNCcnt) staging.
// Shapes: B=512, N=50 (padded to 64), D=256, STEP=3.
// ---------------------------------------------------------------------------

typedef __attribute__((ext_vector_type(16))) _Float16 v16h;
typedef __attribute__((ext_vector_type(8)))  float    v8f;

#define BB    512
#define NN    50
#define DD    256
#define NP    64       // N padded to 64 for 16x16 tiling
#define HSTR  264      // LDS row stride (halves) for h tile (528B = 33*16)
#define ASTR  72       // LDS row stride (halves) for alpha tile
#define MROWS (BB*NN)  // 25600
#define TOT   ((size_t)BB*NN*DD)  // 6,553,600

#define WMMA_F32_F16(a, b, c) \
  __builtin_amdgcn_wmma_f32_16x16x32_f16(false, (a), false, (b), (short)0, (c), false, false)

// Async DMA: global memory -> LDS, 16 bytes per lane, tracked by ASYNCcnt.
__device__ __forceinline__ void async_copy_b128(const _Float16* gsrc, _Float16* ldst) {
  unsigned lds_off = (unsigned)(size_t)ldst;               // LDS byte offset
  unsigned long long gaddr = (unsigned long long)(size_t)gsrc;
  asm volatile("global_load_async_to_lds_b128 %0, %1, off"
               :: "v"(lds_off), "v"(gaddr) : "memory");
}
__device__ __forceinline__ void wait_async0() {
  asm volatile("s_wait_asynccnt 0" ::: "memory");
}

// ---------------------------------------------------------------------------
// Kernel 1: embedding gather -> h, h0 (f32) + h16, h0_16 (f16); zero hsum
// ---------------------------------------------------------------------------
__global__ __launch_bounds__(256) void gather_init_kernel(
    const int* __restrict__ inputs, const float* __restrict__ emb,
    float* __restrict__ h, float* __restrict__ h0, float* __restrict__ hsum,
    _Float16* __restrict__ h16, _Float16* __restrict__ h016) {
  size_t idx = (size_t)blockIdx.x * 256 + threadIdx.x;
  if (idx >= TOT) return;
  size_t row = idx >> 8;         // b*N + n
  int d = (int)(idx & 255);
  int node = inputs[row];
  float v = emb[(size_t)node * DD + d];
  h[idx] = v;
  h0[idx] = v;
  hsum[idx] = 0.0f;
  h16[idx] = (_Float16)v;
  h016[idx] = (_Float16)v;
}

// ---------------------------------------------------------------------------
// Kernel 2: star = masked average pool of h0 over N
// ---------------------------------------------------------------------------
__global__ __launch_bounds__(256) void star_init_kernel(
    const float* __restrict__ h0, const int* __restrict__ gm,
    float* __restrict__ star) {
  int b = blockIdx.x, d = threadIdx.x;
  float s = 0.0f, len = 0.0f;
  for (int n = 0; n < NN; ++n) {
    float g = (float)gm[b * NN + n];
    s += h0[((size_t)b * NN + n) * DD + d] * g;
    len += g;
  }
  star[(size_t)b * DD + d] = s / len;
}

// ---------------------------------------------------------------------------
// Kernel 2b: f32 -> f16 conversion with scale (for hg1 = hsum/3 and weights)
// ---------------------------------------------------------------------------
__global__ __launch_bounds__(256) void cvt_f16_kernel(
    const float* __restrict__ src, _Float16* __restrict__ dst,
    float scale, size_t count) {
  size_t idx = (size_t)blockIdx.x * 256 + threadIdx.x;
  if (idx < count) dst[idx] = (_Float16)(src[idx] * scale);
}

// ---------------------------------------------------------------------------
// Kernel 3: one fused star-GAT step, block-per-batch, 8 waves.
//   async-DMA h16 tile -> LDS, scores (4x WMMA) -> masked softmax ->
//   alpha@h (WMMA) -> star gate mix -> noise -> hsum -> attn pool -> star
// ---------------------------------------------------------------------------
struct __align__(16) GatSmem {
  _Float16 hA[NP * HSTR];     // h[b] as f16, rows >= 50 zero       (33.8 KB)
  _Float16 alp[NP * ASTR];    // softmax alpha as f16               ( 9.2 KB)
  float    aL[4 * DD];        // a0..a3                             ( 4.0 KB)
  float    starL[DD];         // star[b]                            ( 1.0 KB)
  float    work[NP * DD];     // scores 4x64x64 then h_new 64x256   (64.0 KB)
  float    rowA[NP];          // per-row scratch (gate / attn expw)
  float    rowB[NP];          // per-row scratch (noise norm)
  float    sS;                // attn denom
};

__global__ __launch_bounds__(256) void gat_step_kernel(
    float* __restrict__ h_g, _Float16* __restrict__ h16_g,
    const int* __restrict__ A_g,
    const int* __restrict__ gm_g, const float* __restrict__ noise_g,
    const float* __restrict__ a0, const float* __restrict__ a1,
    const float* __restrict__ a2, const float* __restrict__ a3,
    float* __restrict__ star_g, float* __restrict__ hsum_g) {
  __shared__ GatSmem sm;
  const int b = blockIdx.x, tid = threadIdx.x;
  const int wave = tid >> 5, lane = tid & 31;
  const int lmod = lane & 15, lhi = lane >> 4;

  // ---- Phase A: async DMA h16[b] tile into LDS; load a0..a3, star --------
  {
    // 50 rows * 32 chunks of 8 halves (16B) = 1600 chunks
    for (int q = tid; q < NN * 32; q += 256) {
      int r = q >> 5, c8 = (q & 31) * 8;
      async_copy_b128(&h16_g[((size_t)b * NN + r) * DD + c8],
                      &sm.hA[r * HSTR + c8]);
    }
    // zero the pad rows (50..63)
    for (int q = tid; q < (NP - NN) * DD; q += 256) {
      int r = NN + (q >> 8), d = q & 255;
      sm.hA[r * HSTR + d] = (_Float16)0.0f;
    }
    const float* as[4] = {a0, a1, a2, a3};
    for (int k4 = 0; k4 < 4; ++k4) sm.aL[k4 * DD + tid] = as[k4][tid];
    sm.starL[tid] = star_g[(size_t)b * DD + tid];
    wait_async0();
  }
  __syncthreads();

  // ---- Phase B: e_k = leakyrelu((h*a_k) @ h^T), 4 x 64x64, WMMA ----------
  for (int t = wave; t < 64; t += 8) {
    int k4 = t >> 4, it = (t >> 2) & 3, jt = t & 3;
    v8f acc = {};
    for (int kk = 0; kk < 8; ++kk) {
      int k0 = kk * 32;
      v16h af, bf;
      int m = it * 16 + lmod;
#pragma unroll
      for (int v = 0; v < 8; ++v) {
        int k = k0 + (v & 3) * 2 + (v >> 2) * 16 + lhi * 8;
        af[2 * v]     = (_Float16)((float)sm.hA[m * HSTR + k]     * sm.aL[k4 * DD + k]);
        af[2 * v + 1] = (_Float16)((float)sm.hA[m * HSTR + k + 1] * sm.aL[k4 * DD + k + 1]);
      }
      int nj = jt * 16 + lmod;   // B[k=d][n=j] = h[j][d], n = lane%16, k in table order
#pragma unroll
      for (int v = 0; v < 8; ++v) {
        int k = k0 + (v & 3) * 2 + (v >> 2) * 16 + lhi * 8;
        bf[2 * v]     = sm.hA[nj * HSTR + k];
        bf[2 * v + 1] = sm.hA[nj * HSTR + k + 1];
      }
      acc = WMMA_F32_F16(af, bf, acc);
    }
#pragma unroll
    for (int v = 0; v < 8; ++v) {
      int i = it * 16 + v + lhi * 8;
      int j = jt * 16 + lmod;
      float x = acc[v];
      sm.work[(k4 * 64 + i) * 64 + j] = (x > 0.0f) ? x : 0.2f * x;
    }
  }
  __syncthreads();

  // ---- Phase C: select by A, softmax over j, write f16 alpha -------------
  if (tid < 64) {
    int i = tid;
    for (int j = 0; j < 64; ++j) sm.alp[i * ASTR + j] = (_Float16)0.0f;
    if (i < NN) {
      const int* Arow = A_g + ((size_t)b * NN + i) * NN;
      float mx = -3.0e38f;
      for (int j = 0; j < NN; ++j) {
        int av = Arow[j];
        float e = (av >= 1 && av <= 4) ? sm.work[((av - 1) * 64 + i) * 64 + j] : -9.0e15f;
        mx = fmaxf(mx, e);
      }
      float sum = 0.0f;
      for (int j = 0; j < NN; ++j) {
        int av = Arow[j];
        float e = (av >= 1 && av <= 4) ? sm.work[((av - 1) * 64 + i) * 64 + j] : -9.0e15f;
        sum += __expf(e - mx);
      }
      float inv = 1.0f / sum;
      for (int j = 0; j < NN; ++j) {
        int av = Arow[j];
        float e = (av >= 1 && av <= 4) ? sm.work[((av - 1) * 64 + i) * 64 + j] : -9.0e15f;
        sm.alp[i * ASTR + j] = (_Float16)(__expf(e - mx) * inv);
      }
    }
  }
  __syncthreads();

  // ---- Phase D: h_new = alpha @ h  (64x256, K=64), WMMA ------------------
  for (int t = wave; t < 64; t += 8) {
    int it = t >> 4, nt = t & 15;
    v8f acc = {};
#pragma unroll
    for (int kk = 0; kk < 2; ++kk) {
      int k0 = kk * 32;
      v16h af, bf;
      int m = it * 16 + lmod;
#pragma unroll
      for (int v = 0; v < 8; ++v) {
        int k = k0 + (v & 3) * 2 + (v >> 2) * 16 + lhi * 8;
        af[2 * v]     = sm.alp[m * ASTR + k];
        af[2 * v + 1] = sm.alp[m * ASTR + k + 1];
      }
      int nd = nt * 16 + lmod;   // B[k=j][n=d] = h[j][d], n = lane%16, k in table order
#pragma unroll
      for (int v = 0; v < 8; ++v) {
        int k = k0 + (v & 3) * 2 + (v >> 2) * 16 + lhi * 8;
        bf[2 * v]     = sm.hA[k * HSTR + nd];
        bf[2 * v + 1] = sm.hA[(k + 1) * HSTR + nd];
      }
      acc = WMMA_F32_F16(af, bf, acc);
    }
#pragma unroll
    for (int v = 0; v < 8; ++v) {
      int i = it * 16 + v + lhi * 8;
      int d = nt * 16 + lmod;
      sm.work[i * DD + d] = acc[v];   // reuse score buffer as h_new
    }
  }
  __syncthreads();

  // ---- Phase E: star gate mix + normalized noise + hsum ------------------
  if (tid < 64) {
    int i = tid;
    float sim = 0.0f, nn2 = 0.0f;
    if (i < NN) {
      const float* np = noise_g + ((size_t)b * NN + i) * DD;
      for (int d = 0; d < DD; ++d) {
        sim += sm.work[i * DD + d] * sm.starL[d];
        float r = np[d];
        nn2 += r * r;
      }
    }
    sm.rowA[i] = 1.0f / (1.0f + __expf(-sim * 0.0625f));   // sigmoid(sim/16)
    sm.rowB[i] = fmaxf(sqrtf(nn2), 1e-12f);
  }
  __syncthreads();
  for (int idx = tid; idx < NN * DD; idx += 256) {
    int i = idx >> 8, d = idx & 255;
    float ag = sm.rowA[i];
    float x = (1.0f - ag) * sm.work[idx] + ag * sm.starL[d];
    float rn = noise_g[((size_t)b * NN + i) * DD + d] / sm.rowB[i];
    float sg = (x > 0.0f) ? 1.0f : ((x < 0.0f) ? -1.0f : 0.0f);
    x += sg * rn * 0.4f;
    sm.work[idx] = x;
    size_t g = ((size_t)b * NN + i) * DD + d;
    h_g[g] = x;
    h16_g[g] = (_Float16)x;
    hsum_g[g] += x;
  }
  __syncthreads();

  // ---- Phase F: attention pool -> new star -------------------------------
  if (tid < 64) {
    int i = tid;
    float e = 0.0f;
    if (i < NN) {
      float s = 0.0f;
      for (int d = 0; d < DD; ++d) s += sm.work[i * DD + d] * sm.starL[d];
      e = __expf(s) * (float)gm_g[b * NN + i];
    }
    sm.rowA[i] = e;
  }
  __syncthreads();
  if (tid == 0) {
    float s = 0.0f;
    for (int i = 0; i < NN; ++i) s += sm.rowA[i];
    sm.sS = s + 1e-24f;
  }
  __syncthreads();
  {
    int d = tid;
    float acc = 0.0f;
    float inv = 1.0f / sm.sS;
    for (int i = 0; i < NN; ++i) acc += (sm.rowA[i] * inv) * sm.work[i * DD + d];
    star_g[(size_t)b * DD + d] = acc;
  }
}

// ---------------------------------------------------------------------------
// Kernel 4: gate GEMM: a = sigmoid([h0, x2] @ W + b); fused gating epilogue.
//   Tiles async-DMA'd from pre-converted f16 mirrors (hg1_16 pre-scaled).
//   mode 0: alpha_ws = a; out = a*h0 + (1-a)*x2
//   mode 1: out = alpha_ws*h0 + (1-a)*x2*scale
//   M=25600, K=512, Nout=256, tile 64x64, 8 waves x 2 tiles, K-step 32.
// ---------------------------------------------------------------------------
__global__ __launch_bounds__(256) void gate_gemm_kernel(
    const float* __restrict__ h0, const float* __restrict__ x2, float x2scale,
    const _Float16* __restrict__ h016, const _Float16* __restrict__ x16,
    const _Float16* __restrict__ w16, const float* __restrict__ bias,
    float* __restrict__ alpha_ws, float* __restrict__ out, int mode) {
  __shared__ __align__(16) _Float16 At[64 * 40];   // stride 80B = 5*16
  __shared__ __align__(16) _Float16 Wt[32 * 72];   // stride 144B = 9*16
  const int tid = threadIdx.x, wave = tid >> 5, lane = tid & 31;
  const int lmod = lane & 15, lhi = lane >> 4;
  const int bm = blockIdx.x, bn = blockIdx.y;
  v8f acc0 = {}, acc1 = {};

  for (int ks = 0; ks < 16; ++ks) {
    int k0 = ks * 32;
    {   // A tile 64x32 halves: 4 chunks/row -> 256 chunks, 1 per thread
      const _Float16* sp = (k0 < DD) ? h016 : x16;
      int kof = (k0 < DD) ? k0 : (k0 - DD);
      int r = tid >> 2, c8 = (tid & 3) * 8;
      async_copy_b128(&sp[((size_t)bm * 64 + r) * DD + kof + c8],
                      &At[r * 40 + c8]);
    }
    {   // W tile 32x64 halves: 8 chunks/row -> 256 chunks, 1 per thread
      int r = tid >> 3, c8 = (tid & 7) * 8;
      async_copy_b128(&w16[(size_t)(k0 + r) * DD + bn * 64 + c8],
                      &Wt[r * 72 + c8]);
    }
    if (ks < 15)   // prefetch next W tile block
      __builtin_prefetch(&w16[(size_t)(k0 + 32) * DD + bn * 64], 0, 1);
    wait_async0();
    __syncthreads();
#pragma unroll
    for (int u = 0; u < 2; ++u) {
      int tt = wave + u * 8;
      int ti = tt >> 2, tj = tt & 3;
      v16h af, bf;
      int m = ti * 16 + lmod;
#pragma unroll
      for (int v = 0; v < 8; ++v) {
        int k = (v & 3) * 2 + (v >> 2) * 16 + lhi * 8;
        af[2 * v]     = At[m * 40 + k];
        af[2 * v + 1] = At[m * 40 + k + 1];
      }
#pragma unroll
      for (int v = 0; v < 8; ++v) {
        int k = (v & 3) * 2 + (v >> 2) * 16 + lhi * 8;
        bf[2 * v]     = Wt[k * 72 + tj * 16 + lmod];
        bf[2 * v + 1] = Wt[(k + 1) * 72 + tj * 16 + lmod];
      }
      if (u == 0) acc0 = WMMA_F32_F16(af, bf, acc0);
      else        acc1 = WMMA_F32_F16(af, bf, acc1);
    }
    __syncthreads();
  }

#pragma unroll
  for (int u = 0; u < 2; ++u) {
    int tt = wave + u * 8;
    int ti = tt >> 2, tj = tt & 3;
#pragma unroll
    for (int v = 0; v < 8; ++v) {
      int ml = ti * 16 + v + lhi * 8;
      int nl = tj * 16 + lmod;
      size_t m = (size_t)bm * 64 + ml;
      int n = bn * 64 + nl;
      float c = (u == 0 ? acc0[v] : acc1[v]) + bias[n];
      float a = 1.0f / (1.0f + __expf(-c));
      size_t g = m * DD + n;
      float h0v = h0[g];
      float xv = x2[g] * x2scale;
      if (mode == 0) {
        alpha_ws[g] = a;
        out[g] = a * h0v + (1.0f - a) * xv;
      } else {
        out[g] = alpha_ws[g] * h0v + (1.0f - a) * xv;
      }
    }
  }
}

// ---------------------------------------------------------------------------
// Host entry
// ---------------------------------------------------------------------------
extern "C" void kernel_launch(void* const* d_in, const int* in_sizes, int n_in,
                              void* d_out, int out_size, void* d_ws, size_t ws_size,
                              hipStream_t stream) {
  const int*   inputs = (const int*)d_in[0];
  const int*   A      = (const int*)d_in[1];
  const int*   gm     = (const int*)d_in[2];
  // d_in[3] = item (unused by the reference)
  const float* noise  = (const float*)d_in[4];
  const float* emb    = (const float*)d_in[5];
  const float* a0     = (const float*)d_in[6];
  const float* a1     = (const float*)d_in[7];
  const float* a2     = (const float*)d_in[8];
  const float* a3     = (const float*)d_in[9];
  const float* W_hn   = (const float*)d_in[10];
  const float* b_hn   = (const float*)d_in[11];
  const float* W_hn1  = (const float*)d_in[12];
  const float* b_hn1  = (const float*)d_in[13];

  float* out = (float*)d_out;
  float* ws  = (float*)d_ws;

  // f32 workspace
  float* h      = ws;                 // [B,N,D]
  float* h0     = ws + TOT;           // [B,N,D]
  float* hsum   = ws + 2 * TOT;       // [B,N,D]  (sum over steps; hg1 = hsum/3)
  float* alphaW = ws + 3 * TOT;       // [B,N,D]  gate alpha scratch
  float* star   = ws + 4 * TOT;       // [B,D]
  // f16 mirrors (for async-DMA + WMMA staging)
  _Float16* f16base = (_Float16*)(ws + 4 * TOT + (size_t)BB * DD);
  _Float16* h16   = f16base;                    // [B,N,D]
  _Float16* h016  = f16base + TOT;              // [B,N,D]
  _Float16* hg116 = f16base + 2 * TOT;          // [B,N,D]  pre-scaled hsum/3
  _Float16* w16a  = f16base + 3 * TOT;          // [2D, D]
  _Float16* w16b  = w16a + (size_t)2 * DD * DD; // [2D, D]

  float* out0    = out;                         // output  [B,N,D]
  float* outStar = out + TOT;                   // star    [B,D]
  float* out1    = out + TOT + (size_t)BB * DD; // output1 [B,N,D]

  gather_init_kernel<<<(int)((TOT + 255) / 256), 256, 0, stream>>>(
      inputs, emb, h, h0, hsum, h16, h016);
  star_init_kernel<<<BB, 256, 0, stream>>>(h0, gm, star);

  for (int s = 0; s < 3; ++s) {
    gat_step_kernel<<<BB, 256, 0, stream>>>(h, h16, A, gm, noise + (size_t)s * TOT,
                                            a0, a1, a2, a3, star, hsum);
  }

  // f16 mirrors for gate GEMM operands
  size_t wcnt = (size_t)2 * DD * DD;  // 131072
  cvt_f16_kernel<<<(int)((TOT + 255) / 256), 256, 0, stream>>>(hsum, hg116, 1.0f / 3.0f, TOT);
  cvt_f16_kernel<<<(int)((wcnt + 255) / 256), 256, 0, stream>>>(W_hn, w16a, 1.0f, wcnt);
  cvt_f16_kernel<<<(int)((wcnt + 255) / 256), 256, 0, stream>>>(W_hn1, w16b, 1.0f, wcnt);

  dim3 gg(MROWS / 64, DD / 64);
  gate_gemm_kernel<<<gg, 256, 0, stream>>>(h0, h, 1.0f, h016, h16, w16a,
                                           b_hn, alphaW, out0, 0);
  gate_gemm_kernel<<<gg, 256, 0, stream>>>(h0, hsum, 1.0f / 3.0f, h016, hg116, w16b,
                                           b_hn1, alphaW, out1, 1);

  hipMemcpyAsync(outStar, star, (size_t)BB * DD * sizeof(float),
                 hipMemcpyDeviceToDevice, stream);
}
